// DSS_52166672777294
// MI455X (gfx1250) — matrix operand order repaired and checked by reading, not compiled, not run on hardware
//
#include <hip/hip_runtime.h>
#include <hip/hip_bf16.h>

// ---------------------------------------------------------------------------
// Problem constants (from the reference)
// ---------------------------------------------------------------------------
#define T_DIM   32768
#define C_DIM   768
#define NC      7          // n_classes
#define NW      16
#define HALF    8
#define NWP1    17         // NW+1
#define NCOL    24         // NC + NWP1 combined logit columns
#define KTOP    256        // min(NTOKEN, T)

// Workspace layout (bytes)
#define FRAG_OFF   0u
#define FRAG_U32   (24 * 2 * 32 * 8)                 // 12288 u32 = 48 KB of bf16 B-frags
#define GL_OFF     (FRAG_U32 * 4)                    // T x 24 f32 logits
#define GL_BYTES   (T_DIM * NCOL * 4)
#define S_OFF      (GL_OFF + GL_BYTES)               // T x 8 f32 (7 used)
#define S_BYTES    (T_DIM * 8 * 4)
#define PH_OFF     (S_OFF + S_BYTES)                 // T f32 phase
#define PH_BYTES   (T_DIM * 4)
#define IDX_OFF    (PH_OFF + PH_BYTES)               // 256 int32 top-k indices

typedef __attribute__((ext_vector_type(8)))  float  v8f;
typedef __attribute__((ext_vector_type(16))) __bf16 v16bf;

union Frag16 {
    unsigned int u[8];
    uint4        q[2];
    v16bf        v;
};

// Precise RNE f32->bf16 (used once for the tiny weight pre-pack)
__device__ __forceinline__ unsigned int bf16_bits_rne(float f) {
    unsigned int x = __float_as_uint(f);
    x += 0x7FFFu + ((x >> 16) & 1u);
    return (x >> 16) & 0xFFFFu;
}
__device__ __forceinline__ unsigned int pk_bf16_rne(float lo, float hi) {
    return bf16_bits_rne(lo) | (bf16_bits_rne(hi) << 16);
}

// Fast packed f32x2 -> bf16x2: round-half-up bias + one v_perm_b32 that
// extracts the two high halves.  3 VALU ops per packed pair.
// D.b0 = loRnd.b2 (sel 2), D.b1 = loRnd.b3 (3), D.b2 = hiRnd.b2 (6), D.b3 = hiRnd.b3 (7)
__device__ __forceinline__ unsigned int pk_bf16_fast(float lo, float hi) {
    unsigned int lb = __float_as_uint(lo) + 0x8000u;
    unsigned int hb = __float_as_uint(hi) + 0x8000u;
    return __builtin_amdgcn_perm(hb, lb, 0x07060302u);
}

// ---------------------------------------------------------------------------
// Stage 0: pre-pack weight matrix [768 x 24] (Wg | Wl, zero-padded to 32 cols)
// into per-lane WMMA B fragments (16-bit B layout, 32x16 per N-tile).
// Fragment element i of lane l:  K = kk*32 + (l<16 ? 0 : 8) + (i<8 ? i : 8 + i),  N = nt*16 + (l&15)
// ---------------------------------------------------------------------------
__global__ void __launch_bounds__(256)
build_frags_kernel(const float* __restrict__ Wg, const float* __restrict__ Wl,
                   unsigned int* __restrict__ frag) {
    for (int idx = threadIdx.x; idx < FRAG_U32; idx += 256) {
        int p    = idx & 7;          // packed u32 within fragment (elements 2p, 2p+1)
        int lane = (idx >> 3) & 31;
        int nt   = (idx >> 8) & 1;
        int kk   = idx >> 9;         // K chunk 0..23
        int n    = nt * 16 + (lane & 15);
        int kb   = kk * 32 + ((lane < 16) ? 0 : 8);
        int k0   = kb + ((p < 4) ? 2 * p : 16 + 2 * (p - 4));
        int k1   = k0 + 1;
        float v0 = 0.f, v1 = 0.f;
        if (n < NC) {
            v0 = Wg[k0 * NC + n];
            v1 = Wg[k1 * NC + n];
        } else if (n < NCOL) {
            v0 = Wl[k0 * NWP1 + (n - NC)];
            v1 = Wl[k1 * NWP1 + (n - NC)];
        }
        frag[idx] = pk_bf16_rne(v0, v1);
    }
}

// ---------------------------------------------------------------------------
// Stage 1: streaming GEMM. One wave per 16-row tile of X (T/16 = 2048 tiles),
// 8 waves per block. Per K-chunk of 32: build bf16 A-frag from 4 float4 loads
// (the four 32B sectors of each 128B line are consumed exactly once across the
// lane pairs), two v_wmma_f32_16x16x32_bf16 (N-tiles [0,16) and [16,32)).
// Output: GL[t][0..6] = g logits, GL[t][7..23] = l logits (bias added).
// ---------------------------------------------------------------------------
__global__ void __launch_bounds__(256)
gemm_kernel(const float* __restrict__ X, const uint4* __restrict__ fragq,
            const float* __restrict__ bg, const float* __restrict__ bl,
            float* __restrict__ GL) {
    const int lane = threadIdx.x & 31;
    const int wave = threadIdx.x >> 5;
    const int tile = blockIdx.x * 8 + wave;          // 0..2047
    const int row0 = tile * 16;

    const int arow  = row0 + (lane & 15);
    const int kboff = (lane >= 16) ? 8 : 0;          // A-layout half-wave K offset
    const float* xrow = X + (size_t)arow * C_DIM;

    v8f acc0 = {0.f, 0.f, 0.f, 0.f, 0.f, 0.f, 0.f, 0.f};
    v8f acc1 = {0.f, 0.f, 0.f, 0.f, 0.f, 0.f, 0.f, 0.f};

    for (int kk = 0; kk < C_DIM / 32; ++kk) {
        const int col = kk * 32 + kboff;

        // Warm lines 8 chunks ahead within this row (in-bounds by the guard);
        // lowers to global_prefetch_b8.
        if (kk + 8 < C_DIM / 32)
            __builtin_prefetch(xrow + col + 256, 0, 0);

        float4 f0 = *(const float4*)(xrow + col + 0);
        float4 f1 = *(const float4*)(xrow + col + 4);
        float4 f2 = *(const float4*)(xrow + col + 16);
        float4 f3 = *(const float4*)(xrow + col + 20);

        Frag16 a;
        a.u[0] = pk_bf16_fast(f0.x, f0.y);  a.u[1] = pk_bf16_fast(f0.z, f0.w);
        a.u[2] = pk_bf16_fast(f1.x, f1.y);  a.u[3] = pk_bf16_fast(f1.z, f1.w);
        a.u[4] = pk_bf16_fast(f2.x, f2.y);  a.u[5] = pk_bf16_fast(f2.z, f2.w);
        a.u[6] = pk_bf16_fast(f3.x, f3.y);  a.u[7] = pk_bf16_fast(f3.z, f3.w);

        Frag16 b0, b1;
        const uint4* fp0 = fragq + (size_t)((kk * 2 + 0) * 32 + lane) * 2;
        const uint4* fp1 = fragq + (size_t)((kk * 2 + 1) * 32 + lane) * 2;
        b0.q[0] = fp0[0];  b0.q[1] = fp0[1];
        b1.q[0] = fp1[0];  b1.q[1] = fp1[1];

        acc0 = __builtin_amdgcn_wmma_f32_16x16x32_bf16(
            false, a.v, false, b0.v, (short)0, acc0, false, false);
        acc1 = __builtin_amdgcn_wmma_f32_16x16x32_bf16(
            false, a.v, false, b1.v, (short)0, acc1, false, false);
    }

    // C/D layout: element v -> M = v + (lane<16 ? 0 : 8), N = lane & 15
    const int n0 = lane & 15;
    const int n1 = 16 + n0;
    const float bias0 = (n0 < NC) ? bg[n0] : bl[n0 - NC];
    const float bias1 = (n1 < NCOL) ? bl[n1 - NC] : 0.f;
    const int rbase = row0 + ((lane < 16) ? 0 : 8);

#pragma unroll
    for (int v = 0; v < 8; ++v) {
        const int r = rbase + v;
        GL[(size_t)r * NCOL + n0] = acc0[v] + bias0;
        if (n1 < NCOL) GL[(size_t)r * NCOL + n1] = acc1[v] + bias1;
    }
}

// ---------------------------------------------------------------------------
// Stage 2a: per (t, c) windowed softmax over 17 logits; s[t,c] = sum softmax
// logits[t,c,j] = g[t+j-8, c] (zero-padded) + l[t, j]
// ---------------------------------------------------------------------------
__global__ void __launch_bounds__(256)
softmax_kernel(const float* __restrict__ GL, float* __restrict__ S) {
    const int t = blockIdx.x * 256 + threadIdx.x;

    float lv[NWP1];
#pragma unroll
    for (int j = 0; j < NWP1; ++j) lv[j] = GL[(size_t)t * NCOL + NC + j];

#pragma unroll
    for (int c = 0; c < NC; ++c) {
        float lg[NWP1];
        float m = -3.0e38f;
#pragma unroll
        for (int j = 0; j < NWP1; ++j) {
            const int tt = t + j - HALF;
            float g = (tt >= 0 && tt < T_DIM) ? GL[(size_t)tt * NCOL + c] : 0.f;
            lg[j] = g + lv[j];
            m = fmaxf(m, lg[j]);
        }
        float e[NWP1];
        float denom = 0.f;
#pragma unroll
        for (int j = 0; j < NWP1; ++j) { e[j] = __expf(lg[j] - m); denom += e[j]; }
        float s = 0.f;
#pragma unroll
        for (int j = 0; j < NWP1; ++j) {
            float p = e[j] / denom;
            p = (p == p) ? p : 0.f;          // NaN -> 0, as in reference
            s += p;
        }
        S[(size_t)t * 8 + c] = s;
    }
}

// ---------------------------------------------------------------------------
// Stage 2b: 16-wide zero-padded temporal box sum, max over classes -> phase[t]
// final[t,c] = sum_{dt=-8..7} s[t+dt, c]
// ---------------------------------------------------------------------------
__global__ void __launch_bounds__(256)
phase_kernel(const float* __restrict__ S, float* __restrict__ phase) {
    const int t = blockIdx.x * 256 + threadIdx.x;
    float ph = -3.0e38f;
#pragma unroll
    for (int c = 0; c < NC; ++c) {
        float sum = 0.f;
#pragma unroll
        for (int w = 0; w < NW; ++w) {
            const int tt = t + w - HALF;
            if (tt >= 0 && tt < T_DIM) sum += S[(size_t)tt * 8 + c];
        }
        ph = fmaxf(ph, sum);
    }
    phase[t] = ph;
}

// ---------------------------------------------------------------------------
// Stage 3: stable top-256 of phase[0..T).  Key = orderable(phase)<<32 | ~t
// so u64-max reproduces jax.lax.top_k ordering (ties -> lowest index).
// Single block of 1024 threads, 32 keys/thread in registers, shuffle reduce.
// ---------------------------------------------------------------------------
__global__ void __launch_bounds__(1024)
topk_kernel(const float* __restrict__ phase, int* __restrict__ idx_out) {
    const int tid  = threadIdx.x;
    const int lane = tid & 31;
    const int wid  = tid >> 5;

    unsigned long long key[32];
#pragma unroll
    for (int j = 0; j < 32; ++j) {
        const int t = tid * 32 + j;
        unsigned int u = __float_as_uint(phase[t]);
        u = (u & 0x80000000u) ? ~u : (u | 0x80000000u);
        key[j] = ((unsigned long long)u << 32) | (unsigned int)(~t);
    }

    __shared__ unsigned long long red[32];
    __shared__ unsigned long long winner_s;

    for (int it = 0; it < KTOP; ++it) {
        unsigned long long m = 0ull;
#pragma unroll
        for (int j = 0; j < 32; ++j) m = (key[j] > m) ? key[j] : m;
#pragma unroll
        for (int off = 16; off > 0; off >>= 1) {
            unsigned long long o = __shfl_xor(m, off, 32);
            m = (o > m) ? o : m;
        }
        if (lane == 0) red[wid] = m;
        __syncthreads();
        if (wid == 0) {
            unsigned long long m2 = red[lane];
#pragma unroll
            for (int off = 16; off > 0; off >>= 1) {
                unsigned long long o = __shfl_xor(m2, off, 32);
                m2 = (o > m2) ? o : m2;
            }
            if (lane == 0) {
                winner_s = m2;
                idx_out[it] = (int)(~(unsigned int)(m2 & 0xFFFFFFFFull));
            }
        }
        __syncthreads();
        const unsigned long long w = winner_s;
#pragma unroll
        for (int j = 0; j < 32; ++j)
            if (key[j] == w) key[j] = 0ull;
        // next iteration's first __syncthreads orders these reads vs. the
        // next winner_s write
    }
}

// ---------------------------------------------------------------------------
// Stage 4: gather the 256 winning rows of frame_feature; append indices.
// d_out = [ gathered (256*768 f32) | indices (256, as f32) ]
// ---------------------------------------------------------------------------
__global__ void __launch_bounds__(256)
gather_kernel(const float* __restrict__ X, const int* __restrict__ idx,
              float* __restrict__ out) {
    const int b = blockIdx.x;
    const int t = idx[b];
    const float* src = X + (size_t)t * C_DIM;
    float* dst = out + (size_t)b * C_DIM;
    for (int c = threadIdx.x; c < C_DIM; c += 256) dst[c] = src[c];
    if (threadIdx.x == 0) out[(size_t)KTOP * C_DIM + b] = (float)t;
}

// ---------------------------------------------------------------------------
extern "C" void kernel_launch(void* const* d_in, const int* in_sizes, int n_in,
                              void* d_out, int out_size, void* d_ws, size_t ws_size,
                              hipStream_t stream) {
    (void)in_sizes; (void)n_in; (void)out_size; (void)ws_size;

    const float* frame = (const float*)d_in[0];   // (T,1,768) f32
    const float* Wg    = (const float*)d_in[1];   // (768,7)
    const float* bg    = (const float*)d_in[2];   // (7,)
    const float* Wl    = (const float*)d_in[3];   // (768,17)
    const float* bl    = (const float*)d_in[4];   // (17,)

    char* ws = (char*)d_ws;
    unsigned int* frag  = (unsigned int*)(ws + FRAG_OFF);
    float*        GL    = (float*)(ws + GL_OFF);
    float*        S     = (float*)(ws + S_OFF);
    float*        phase = (float*)(ws + PH_OFF);
    int*          idx   = (int*)(ws + IDX_OFF);
    float*        out   = (float*)d_out;

    build_frags_kernel<<<1, 256, 0, stream>>>(Wg, Wl, frag);
    gemm_kernel<<<T_DIM / 16 / 8, 256, 0, stream>>>(frame, (const uint4*)frag,
                                                    bg, bl, GL);
    softmax_kernel<<<T_DIM / 256, 256, 0, stream>>>(GL, S);
    phase_kernel<<<T_DIM / 256, 256, 0, stream>>>(S, phase);
    topk_kernel<<<1, 1024, 0, stream>>>(phase, idx);
    gather_kernel<<<KTOP, 256, 0, stream>>>(frame, idx, out);
}
